// FlaxGPTJBlock_35820027248683
// MI455X (gfx1250) — matrix operand order repaired and checked by reading, not compiled, not run on hardware
//
#include <hip/hip_runtime.h>
#include <hip/hip_bf16.h>
#include <math.h>

typedef __attribute__((ext_vector_type(16))) __bf16        v16bf;
typedef __attribute__((ext_vector_type(8)))  float         v8f;
typedef __attribute__((ext_vector_type(4)))  float         v4f;
typedef __attribute__((ext_vector_type(4)))  unsigned int  v4u;
typedef int v4i_nat __attribute__((vector_size(16)));

#define B_      2
#define S_      2048
#define H_      4096
#define NH_     16
#define HD_     256
#define RD_     64
#define INNER_  16384
#define TOKENS  (B_ * S_)

union FragAB { v16bf v; v4u u[2]; };
union Cvt16  { __bf16 h[16]; v4u u[2]; };

// ---- CDNA5 async global->LDS copy path (ASYNCcnt), with safe fallback ----
#if defined(__has_builtin)
#if __has_builtin(__builtin_amdgcn_global_load_async_to_lds_b128)
#define USE_ASYNC_LDS 1
#endif
#endif
#ifndef USE_ASYNC_LDS
#define USE_ASYNC_LDS 0
#endif

#define AS1 __attribute__((address_space(1)))
#define AS3 __attribute__((address_space(3)))

#if USE_ASYNC_LDS
__device__ __forceinline__ void async_copy16(const void* g, void* l) {
  __builtin_amdgcn_global_load_async_to_lds_b128(
      (AS1 v4i_nat*)(AS1 void*)(void*)g,
      (AS3 v4i_nat*)(AS3 void*)l, 0, 0);
}
#define ASYNC_WAIT() asm volatile("s_wait_asynccnt 0x0" ::: "memory")
#else
__device__ __forceinline__ void async_copy16(const void* g, void* l) {
  *(v4u*)l = *(const v4u*)g;
}
#define ASYNC_WAIT()
#endif

__device__ __forceinline__ float gelu_tanh(float v) {
  float c = 0.7978845608028654f * (v + 0.044715f * v * v * v);
  return 0.5f * v * (1.0f + tanhf(c));
}

// ---------------- LayerNorm: x f32 [4096,4096] -> h bf16 ----------------
__global__ __launch_bounds__(256) void ln_kernel(
    const float* __restrict__ x, const float* __restrict__ gamma,
    const float* __restrict__ beta, __bf16* __restrict__ h) {
  __shared__ float r1[8], r2[8];
  int row = blockIdx.x;
  int t = threadIdx.x;
  const float* xr = x + (size_t)row * H_ + t * 16;
  float loc[16];
  float s1 = 0.f, s2 = 0.f;
#pragma unroll
  for (int i = 0; i < 4; ++i) {
    v4f v = ((const v4f*)xr)[i];
#pragma unroll
    for (int j = 0; j < 4; ++j) {
      float f = v[j];
      loc[i * 4 + j] = f;
      s1 += f; s2 += f * f;
    }
  }
#pragma unroll
  for (int m = 1; m < 32; m <<= 1) {
    s1 += __shfl_xor(s1, m);
    s2 += __shfl_xor(s2, m);
  }
  int wave = t >> 5, lane = t & 31;
  if (lane == 0) { r1[wave] = s1; r2[wave] = s2; }
  __syncthreads();
  float t1 = 0.f, t2 = 0.f;
#pragma unroll
  for (int i = 0; i < 8; ++i) { t1 += r1[i]; t2 += r2[i]; }
  float mean = t1 * (1.0f / H_);
  float var  = t2 * (1.0f / H_) - mean * mean;
  float rstd = rsqrtf(var + 1e-5f);
  __bf16* hr = h + (size_t)row * H_ + t * 16;
#pragma unroll
  for (int i = 0; i < 16; ++i) {
    int c = t * 16 + i;
    hr[i] = (__bf16)((loc[i] - mean) * rstd * gamma[c] + beta[c]);
  }
}

// ---------------- Double-buffered WMMA GEMM: C[M,N] = A_bf16[M,K] @ W_f32[K,N] ------
// Block 128x128, 8 waves, wave tile 32x64, K-step 32. A-tiles go global->LDS via the
// CDNA5 async path; f32 weights are converted to bf16 in registers (read from HBM once).
template<bool BIAS, bool GELU, bool OUT_BF16, bool FINAL>
__global__ __launch_bounds__(256) void gemm_bf16_kernel(
    const __bf16* __restrict__ A, const float* __restrict__ W,
    const float* __restrict__ bias, const float* __restrict__ addA,
    const float* __restrict__ addB, __bf16* __restrict__ outb,
    float* __restrict__ outf, int N, int K) {
  __shared__ __align__(16) __bf16 sA[2][128 * 32];
  __shared__ __align__(16) __bf16 sW[2][32 * 128];
  const int ABUF = 128 * 32, WBUF = 32 * 128;
  int tid  = threadIdx.x;
  int wave = tid >> 5, lane = tid & 31;
  int wm = wave & 3, wn = wave >> 2;
  int lhalf = lane >> 4, lrow = lane & 15;
  int m0 = blockIdx.y * 128, n0 = blockIdx.x * 128;

  v8f acc[2][4] = {};

  int arow = tid >> 1;
  int acol = (tid & 1) * 16;
  const __bf16* Aptr = A + (size_t)(m0 + arow) * K + acol;
  int wrow = tid >> 3;
  int wcol = (tid & 7) * 16;
  const float* Wptr = W + (size_t)wrow * N + n0 + wcol;

  __bf16* sAdst = &sA[0][arow * 32 + acol];
  __bf16* sWdst = &sW[0][wrow * 128 + wcol];

  v4f wreg[4];
#if !USE_ASYNC_LDS
  v4u areg0, areg1;
#endif

  // ---- stage tile 0 ----
  async_copy16(Aptr, sAdst);
  async_copy16(Aptr + 8, sAdst + 8);
#pragma unroll
  for (int i = 0; i < 4; ++i) wreg[i] = ((const v4f*)Wptr)[i];
  {
    Cvt16 wc;
#pragma unroll
    for (int i = 0; i < 4; ++i)
#pragma unroll
      for (int j = 0; j < 4; ++j) wc.h[i * 4 + j] = (__bf16)wreg[i][j];
    ((v4u*)sWdst)[0] = wc.u[0];
    ((v4u*)sWdst)[1] = wc.u[1];
  }
  ASYNC_WAIT();
  __syncthreads();

  int nk = K >> 5;
  for (int kt = 0; kt < nk; ++kt) {
    int cur = kt & 1, nxt = cur ^ 1;
    bool hn = (kt + 1) < nk;
    if (hn) {
      const __bf16* An = Aptr + (size_t)(kt + 1) * 32;
      const float*  Wn = Wptr + (size_t)(kt + 1) * 32 * N;
#if USE_ASYNC_LDS
      async_copy16(An, sAdst + nxt * ABUF);
      async_copy16(An + 8, sAdst + nxt * ABUF + 8);
#else
      areg0 = ((const v4u*)An)[0];
      areg1 = ((const v4u*)An)[1];
#endif
#pragma unroll
      for (int i = 0; i < 4; ++i) wreg[i] = ((const v4f*)Wn)[i];
      if (kt + 2 < nk)
        __builtin_prefetch(Wptr + (size_t)(kt + 2) * 32 * N, 0, 3);
    }

    // ---- compute on buffer `cur` ----
    FragAB a[2], b[4];
#pragma unroll
    for (int i = 0; i < 2; ++i) {
      // 16-bit A layout: lanes 0-15 M=0..15, K=[koff..koff+7],[koff+16..koff+23]
      const __bf16* p = &sA[cur][(wm * 32 + i * 16 + lrow) * 32 + lhalf * 8];
      a[i].u[0] = *(const v4u*)p;
      a[i].u[1] = *(const v4u*)(p + 16);
    }
#pragma unroll
    for (int j = 0; j < 4; ++j) {
      // B layout: lane column N = lrow, K = [lhalf*8..+7],[16+lhalf*8..+7]
      const __bf16* p = &sW[cur][wn * 64 + j * 16 + lrow];
#pragma unroll
      for (int s = 0; s < 16; ++s)
        b[j].v[s] = p[((s < 8) ? (lhalf * 8 + s) : (16 + lhalf * 8 + s - 8)) * 128];
    }
#pragma unroll
    for (int i = 0; i < 2; ++i)
#pragma unroll
      for (int j = 0; j < 4; ++j)
        acc[i][j] = __builtin_amdgcn_wmma_f32_16x16x32_bf16(
            false, a[i].v, false, b[j].v, (short)0, acc[i][j], false, false);

    if (hn) {
      Cvt16 wc;
#pragma unroll
      for (int i = 0; i < 4; ++i)
#pragma unroll
        for (int j = 0; j < 4; ++j) wc.h[i * 4 + j] = (__bf16)wreg[i][j];
      ((v4u*)(sWdst + nxt * WBUF))[0] = wc.u[0];
      ((v4u*)(sWdst + nxt * WBUF))[1] = wc.u[1];
#if !USE_ASYNC_LDS
      ((v4u*)(sAdst + nxt * ABUF))[0] = areg0;
      ((v4u*)(sAdst + nxt * ABUF))[1] = areg1;
#endif
    }
    ASYNC_WAIT();
    __syncthreads();
  }

#pragma unroll
  for (int i = 0; i < 2; ++i) {
#pragma unroll
    for (int j = 0; j < 4; ++j) {
      int col = n0 + wn * 64 + j * 16 + lrow;
      float bv = BIAS ? bias[col] : 0.f;
#pragma unroll
      for (int r = 0; r < 8; ++r) {
        int row = m0 + wm * 32 + i * 16 + lhalf * 8 + r;
        float v = acc[i][j][r] + bv;
        if (GELU) v = gelu_tanh(v);
        size_t idx = (size_t)row * N + col;
        if (FINAL) v += addA[idx] + addB[idx];
        if (OUT_BF16) outb[idx] = (__bf16)v;
        else          outf[idx] = v;
      }
    }
  }
}

// ---------------- RoPE (rotate-every-two) + q scaling + K^T materialization ----------------
__global__ __launch_bounds__(256) void rope_kernel(
    __bf16* __restrict__ q, const __bf16* __restrict__ k,
    __bf16* __restrict__ kT, const int* __restrict__ pos_ids) {
  int bs = blockIdx.x;
  int b = bs >> 11, s = bs & 2047;
  float p = (float)pos_ids[bs];
  int t = threadIdx.x;
  int hh = t >> 4;
  int d0 = (t & 15) * 16;
  size_t base = (size_t)bs * H_ + hh * HD_ + d0;
  float qa[16], ka[16];
#pragma unroll
  for (int i = 0; i < 16; ++i) {
    qa[i] = (float)q[base + i];
    ka[i] = (float)k[base + i];
  }
  if (d0 < RD_) {
#pragma unroll
    for (int i = 0; i < 8; ++i) {
      int pi = (d0 >> 1) + i;  // pair index 0..31
      float ang = p * __powf(10000.f, -(float)(2 * pi) * (1.0f / 64.0f));
      float sn, cs;
      __sincosf(ang, &sn, &cs);
      float q0v = qa[2 * i], q1v = qa[2 * i + 1];
      qa[2 * i]     = q0v * cs - q1v * sn;
      qa[2 * i + 1] = q1v * cs + q0v * sn;
      float k0v = ka[2 * i], k1v = ka[2 * i + 1];
      ka[2 * i]     = k0v * cs - k1v * sn;
      ka[2 * i + 1] = k1v * cs + k0v * sn;
    }
  }
#pragma unroll
  for (int i = 0; i < 16; ++i) q[base + i] = (__bf16)(qa[i] * 0.0625f); // 1/sqrt(256)
  size_t kb = ((size_t)(b * NH_ + hh) * HD_ + d0) * S_ + s;
#pragma unroll
  for (int i = 0; i < 16; ++i) kT[kb + (size_t)i * S_] = (__bf16)ka[i];
}

// ---------------- Flash attention: block=(128 q-rows, head, batch), 8 waves x 16 rows ----
__global__ __launch_bounds__(256) void flash_attn_kernel(
    const __bf16* __restrict__ q, const __bf16* __restrict__ kT,
    const __bf16* __restrict__ v, const int* __restrict__ amask,
    __bf16* __restrict__ ctx) {
  extern __shared__ __align__(16) char smem[];
  __bf16* sQ = (__bf16*)smem;                 // 128 x 256       (65536 B)
  __bf16* sK = (__bf16*)(smem + 65536);       // 256(d) x 32(kp) (16384 B)
  __bf16* sV = (__bf16*)(smem + 81920);       // 32(kp) x 256(d) (16384 B)
  __bf16* sP = (__bf16*)(smem + 98304);       // 8 waves x 16x32 (8192 B)
  int*    sM = (int*)(smem + 106496);         // 32 ints

  int tid  = threadIdx.x;
  int wave = tid >> 5, lane = tid & 31;
  int lhalf = lane >> 4, lrow = lane & 15;
  int q0 = blockIdx.x * 128;
  int h  = blockIdx.y;
  int b  = blockIdx.z;

  { // load Q tile once (async path)
    int r = tid >> 1, half = tid & 1;
    const __bf16* src = q + (size_t)(b * S_ + q0 + r) * H_ + h * HD_ + half * 128;
    __bf16* dst = sQ + r * 256 + half * 128;
#pragma unroll
    for (int i = 0; i < 16; ++i) async_copy16(src + i * 8, dst + i * 8);
  }

  v8f cacc[16] = {};
  float mrow[8], lsum[8];
#pragma unroll
  for (int r = 0; r < 8; ++r) { mrow[r] = -3.0e38f; lsum[r] = 0.f; }

  int ntiles = q0 / 32 + 4;  // causal: only keys <= q0+127
  for (int kt = 0; kt < ntiles; ++kt) {
    int c0 = kt * 32;
    __syncthreads();
    { // K^T tile: 256 d-rows x 32 key positions (row-contiguous in kT)
      const __bf16* src = kT + ((size_t)((b * NH_ + h) * HD_ + tid)) * S_ + c0;
      __bf16* dst = sK + tid * 32;
#pragma unroll
      for (int i = 0; i < 4; ++i) async_copy16(src + i * 8, dst + i * 8);
    }
    { // V tile: 32 key rows x 256 d
      int vr = tid >> 3, seg = tid & 7;
      const __bf16* src = v + (size_t)(b * S_ + c0 + vr) * H_ + h * HD_ + seg * 32;
      __bf16* dst = sV + vr * 256 + seg * 32;
#pragma unroll
      for (int i = 0; i < 4; ++i) async_copy16(src + i * 8, dst + i * 8);
    }
    if (tid < 32) sM[tid] = amask[b * S_ + c0 + tid];
    ASYNC_WAIT();
    __syncthreads();

    // scores 16x32 per wave: 8 d-chunks x 2 key sub-tiles
    v8f sc[2] = {};
#pragma unroll
    for (int c = 0; c < 8; ++c) {
      FragAB aq;
      const __bf16* p = &sQ[(wave * 16 + lrow) * 256 + c * 32 + lhalf * 8];
      aq.u[0] = *(const v4u*)p;
      aq.u[1] = *(const v4u*)(p + 16);
#pragma unroll
      for (int t2 = 0; t2 < 2; ++t2) {
        FragAB bk;
        const __bf16* pb = &sK[(c * 32) * 32 + t2 * 16 + lrow];
#pragma unroll
        for (int s = 0; s < 16; ++s)
          bk.v[s] = pb[((s < 8) ? (lhalf * 8 + s) : (16 + lhalf * 8 + s - 8)) * 32];
        sc[t2] = __builtin_amdgcn_wmma_f32_16x16x32_bf16(
            false, aq.v, false, bk.v, (short)0, sc[t2], false, false);
      }
    }

    // causal + padding mask (C layout: VGPR r -> M = r + 8*lhalf, N = lrow)
    int km0 = sM[lrow], km1 = sM[16 + lrow];
#pragma unroll
    for (int t2 = 0; t2 < 2; ++t2) {
      int kpos = c0 + t2 * 16 + lrow;
      int km = t2 ? km1 : km0;
#pragma unroll
      for (int r = 0; r < 8; ++r) {
        int qrow = q0 + wave * 16 + lhalf * 8 + r;
        bool ok = (kpos <= qrow) && (km > 0);
        sc[t2][r] = ok ? sc[t2][r] : -3.0e38f;
      }
    }

    // online softmax; cross-lane row reductions stay within 16-lane halves
#pragma unroll
    for (int r = 0; r < 8; ++r) {
      float m = fmaxf(sc[0][r], sc[1][r]);
      m = fmaxf(m, __shfl_xor(m, 1));
      m = fmaxf(m, __shfl_xor(m, 2));
      m = fmaxf(m, __shfl_xor(m, 4));
      m = fmaxf(m, __shfl_xor(m, 8));
      float mn = fmaxf(mrow[r], m);
      float corr = __expf(mrow[r] - mn);
      mrow[r] = mn;
      lsum[r] *= corr;
#pragma unroll
      for (int n = 0; n < 16; ++n) cacc[n][r] *= corr;
      float p0 = (sc[0][r] > -1.0e37f) ? __expf(sc[0][r] - mn) : 0.f;
      float p1 = (sc[1][r] > -1.0e37f) ? __expf(sc[1][r] - mn) : 0.f;
      sc[0][r] = p0; sc[1][r] = p1;
      float ps = p0 + p1;
      ps += __shfl_xor(ps, 1);
      ps += __shfl_xor(ps, 2);
      ps += __shfl_xor(ps, 4);
      ps += __shfl_xor(ps, 8);
      lsum[r] += ps;
    }

    // P (C layout) -> wave-private LDS -> A-fragment layout
    __bf16* wp = sP + wave * (16 * 32);
#pragma unroll
    for (int t2 = 0; t2 < 2; ++t2)
#pragma unroll
      for (int r = 0; r < 8; ++r)
        wp[(lhalf * 8 + r) * 32 + t2 * 16 + lrow] = (__bf16)sc[t2][r];
    FragAB pa;
    {
      const __bf16* p = &wp[lrow * 32 + lhalf * 8];
      pa.u[0] = *(const v4u*)p;
      pa.u[1] = *(const v4u*)(p + 16);
    }
    // ctx += P @ V : 16 d-subtiles
#pragma unroll
    for (int n = 0; n < 16; ++n) {
      FragAB bv;
      const __bf16* pb = &sV[n * 16 + lrow];
#pragma unroll
      for (int s = 0; s < 16; ++s)
        bv.v[s] = pb[((s < 8) ? (lhalf * 8 + s) : (16 + lhalf * 8 + s - 8)) * 256];
      cacc[n] = __builtin_amdgcn_wmma_f32_16x16x32_bf16(
          false, pa.v, false, bv.v, (short)0, cacc[n], false, false);
    }
  }

  float inv[8];
#pragma unroll
  for (int r = 0; r < 8; ++r) inv[r] = 1.0f / lsum[r];
#pragma unroll
  for (int n = 0; n < 16; ++n) {
#pragma unroll
    for (int r = 0; r < 8; ++r) {
      int qrow = q0 + wave * 16 + lhalf * 8 + r;
      size_t idx = (size_t)(b * S_ + qrow) * H_ + h * HD_ + n * 16 + lrow;
      ctx[idx] = (__bf16)(cacc[n][r] * inv[r]);
    }
  }
}

// ---------------- host side ----------------
extern "C" void kernel_launch(void* const* d_in, const int* in_sizes, int n_in,
                              void* d_out, int out_size, void* d_ws, size_t ws_size,
                              hipStream_t stream) {
  (void)in_sizes; (void)n_in; (void)out_size; (void)ws_size;
  const float* x      = (const float*)d_in[0];
  const int*   amask  = (const int*)d_in[1];
  const int*   posids = (const int*)d_in[2];
  const float* gamma  = (const float*)d_in[3];
  const float* beta   = (const float*)d_in[4];
  const float* wq     = (const float*)d_in[5];
  const float* wk     = (const float*)d_in[6];
  const float* wv     = (const float*)d_in[7];
  const float* wo     = (const float*)d_in[8];
  const float* w_in   = (const float*)d_in[9];
  const float* b_in   = (const float*)d_in[10];
  const float* w_out  = (const float*)d_in[11];
  const float* b_out  = (const float*)d_in[12];
  float* out = (float*)d_out;

  char* ws = (char*)d_ws;
  const size_t MB = 1024 * 1024;
  __bf16* h_b    = (__bf16*)(ws + 0);        // 32 MiB
  __bf16* q_b    = (__bf16*)(ws + 32 * MB);  // 32 MiB
  __bf16* k_b    = (__bf16*)(ws + 64 * MB);  // 32 MiB
  __bf16* v_b    = (__bf16*)(ws + 96 * MB);  // 32 MiB
  __bf16* kT_b   = (__bf16*)(ws + 128 * MB); // 32 MiB
  __bf16* ctx_b  = (__bf16*)(ws + 160 * MB); // 32 MiB
  float*  attn_f = (float*)(ws + 192 * MB);  // 64 MiB
  __bf16* act_b  = (__bf16*)(ws + 32 * MB);  // 128 MiB, reuses q/k/v/kT after attention

  ln_kernel<<<TOKENS, 256, 0, stream>>>(x, gamma, beta, h_b);

  dim3 gH(H_ / 128, TOKENS / 128);
  gemm_bf16_kernel<false, false, true, false><<<gH, 256, 0, stream>>>(
      h_b, wq, nullptr, nullptr, nullptr, q_b, nullptr, H_, H_);
  gemm_bf16_kernel<false, false, true, false><<<gH, 256, 0, stream>>>(
      h_b, wk, nullptr, nullptr, nullptr, k_b, nullptr, H_, H_);
  gemm_bf16_kernel<false, false, true, false><<<gH, 256, 0, stream>>>(
      h_b, wv, nullptr, nullptr, nullptr, v_b, nullptr, H_, H_);

  rope_kernel<<<TOKENS, 256, 0, stream>>>(q_b, k_b, kT_b, posids);

  flash_attn_kernel<<<dim3(S_ / 128, NH_, B_), 256, 106752, stream>>>(
      q_b, kT_b, v_b, amask, ctx_b);

  gemm_bf16_kernel<false, false, false, false><<<gH, 256, 0, stream>>>(
      ctx_b, wo, nullptr, nullptr, nullptr, nullptr, attn_f, H_, H_);

  dim3 gI(INNER_ / 128, TOKENS / 128);
  gemm_bf16_kernel<true, true, true, false><<<gI, 256, 0, stream>>>(
      h_b, w_in, b_in, nullptr, nullptr, act_b, nullptr, INNER_, H_);

  gemm_bf16_kernel<true, false, false, true><<<gH, 256, 0, stream>>>(
      act_b, w_out, b_out, x, attn_f, nullptr, out, H_, INNER_);
}
